// DenseAtt_46832323395926
// MI455X (gfx1250) — matrix operand order repaired and checked by reading, not compiled, not run on hardware
//
#include <hip/hip_runtime.h>

#define ALPHA 0.2f
#define NTOK 1024
#define DD 64
#define WIDTH 64

typedef __attribute__((ext_vector_type(16))) _Float16 v16h;
typedef __attribute__((ext_vector_type(8)))  float    v8f;

__device__ __forceinline__ float lrelu(float v) { return fmaxf(v, ALPHA * v); }

// Kernel 1: hi[i,k] = x[i,:] . W1[k,0:64] ; hjb[j,k] = x[j,:] . W1[k,64:128] + b1[k]
__global__ void prep_kernel(const float* __restrict__ x,
                            const float* __restrict__ W1,
                            const float* __restrict__ b1,
                            float* __restrict__ hi,
                            float* __restrict__ hjb) {
    int idx = blockIdx.x * blockDim.x + threadIdx.x;   // 0 .. 65535
    int i = idx >> 6;
    int k = idx & 63;
    const float* xr = x + i * DD;
    const float* w  = W1 + k * (2 * DD);
    float s1 = 0.f, s2 = 0.f;
#pragma unroll 8
    for (int d = 0; d < DD; ++d) {
        float xv = xr[d];
        s1 += xv * w[d];
        s2 += xv * w[DD + d];
    }
    hi[idx]  = s1;
    hjb[idx] = s2 + b1[k];
}

// One block per softmax group: blockIdx.x = i*4 + r, covering j in [r*256, r*256+256).
// 8 waves x 2 M-tiles x 16 j's = 256 pairs. Fully fused layer2+layer3+softmax.
__global__ void __launch_bounds__(256)
fused_attn_kernel(const float* __restrict__ hi,
                  const float* __restrict__ hjb,
                  const float* __restrict__ W2,
                  const float* __restrict__ b2,
                  const float* __restrict__ W3,
                  const float* __restrict__ b3,
                  float* __restrict__ out) {
    __shared__ float    hi_s[64];
    __shared__ _Float16 a_lds [8][16 * 64];
    __shared__ _Float16 h2_lds[8][16 * 64];
    __shared__ float    s_lds[1024];
    __shared__ float    red_lds[256];

    const int tid  = threadIdx.x;
    const int wave = tid >> 5;
    const int lane = tid & 31;
    const int i    = blockIdx.x >> 2;
    const int r    = blockIdx.x & 3;
    const int j0   = r << 8;

    const int nsel = lane & 15;       // column for B / C-D layouts, row for A gather
    const int hi16 = lane >> 4;       // lane half
    const int akb  = hi16 * 16;       // A-layout K base within 32-slice
    const int bkb  = hi16 * 16;       // B-layout K base within 32-slice

    // --- Build B tiles in registers: W2 (4 N-blocks x 2 K-slices), W3 (2 K-slices, padded) ---
    v16h bW2[4][2];
#pragma unroll
    for (int vb = 0; vb < 4; ++vb) {
#pragma unroll
        for (int kb = 0; kb < 2; ++kb) {
            const float* wrow = W2 + (vb * 16 + nsel) * WIDTH + kb * 32 + bkb;
            v16h t;
#pragma unroll
            for (int e = 0; e < 16; ++e) t[e] = (_Float16)wrow[e];
            bW2[vb][kb] = t;
        }
    }
    v16h bW3[2];
#pragma unroll
    for (int kb = 0; kb < 2; ++kb) {
        v16h t;
        if (nsel < 4) {
            const float* wrow = W3 + nsel * WIDTH + kb * 32 + bkb;
#pragma unroll
            for (int e = 0; e < 16; ++e) t[e] = (_Float16)wrow[e];
        } else {
#pragma unroll
            for (int e = 0; e < 16; ++e) t[e] = (_Float16)0.0f;
        }
        bW3[kb] = t;
    }

    if (tid < 64) hi_s[tid] = hi[i * DD + tid];
    __syncthreads();

    const v8f czero = {};

#pragma unroll
    for (int t = 0; t < 2; ++t) {
        const int mt = wave * 2 + t;          // M-tile index 0..15 within block
        const int jt = j0 + mt * 16;

        // Stage A tile: lane covers row (lane&15), k-half (lane>>4)*32
        {
            const int row = lane & 15;
            const int kh  = hi16 * 32;
            const float* hjr = hjb + (size_t)(jt + row) * WIDTH + kh;
            _Float16* dst = &a_lds[wave][row * 64 + kh];
#pragma unroll
            for (int k = 0; k < 32; ++k)
                dst[k] = (_Float16)lrelu(hi_s[kh + k] + hjr[k]);
        }
        asm volatile("s_wait_dscnt 0" ::: "memory");   // in-wave LDS RAW

        // Gather A regs in contiguous K order (16 contiguous halves, matching B)
        v16h aA[2];
#pragma unroll
        for (int kb = 0; kb < 2; ++kb) {
            const _Float16* p0 = &a_lds[wave][nsel * 64 + kb * 32 + akb];
            v16h t2;
#pragma unroll
            for (int e = 0; e < 8; ++e) { t2[e] = p0[e]; t2[8 + e] = p0[8 + e]; }
            aA[kb] = t2;
        }

        // Layer 2: 16x64 output = 4 WMMA tiles, K=64 split in two
        v8f acc[4];
#pragma unroll
        for (int vb = 0; vb < 4; ++vb) {
            v8f c = __builtin_amdgcn_wmma_f32_16x16x32_f16(false, aA[0], false, bW2[vb][0],
                                                           (short)0, czero, false, false);
            acc[vb] = __builtin_amdgcn_wmma_f32_16x16x32_f16(false, aA[1], false, bW2[vb][1],
                                                             (short)0, c, false, false);
        }

        // bias + lrelu, scatter D (lane=column) into row-major f16 LDS tile
        {
            const int mb = hi16 * 8;
#pragma unroll
            for (int vb = 0; vb < 4; ++vb) {
                const float bias = b2[vb * 16 + nsel];
#pragma unroll
                for (int rr = 0; rr < 8; ++rr)
                    h2_lds[wave][(rr + mb) * 64 + vb * 16 + nsel] =
                        (_Float16)lrelu(acc[vb][rr] + bias);
            }
        }
        asm volatile("s_wait_dscnt 0" ::: "memory");

        // Gather layer-3 A
        v16h a3[2];
#pragma unroll
        for (int kb = 0; kb < 2; ++kb) {
            const _Float16* p0 = &h2_lds[wave][nsel * 64 + kb * 32 + akb];
            v16h t2;
#pragma unroll
            for (int e = 0; e < 8; ++e) { t2[e] = p0[e]; t2[8 + e] = p0[8 + e]; }
            a3[kb] = t2;
        }

        // Layer 3: heads (4 valid columns of 16)
        v8f sacc = __builtin_amdgcn_wmma_f32_16x16x32_f16(false, a3[0], false, bW3[0],
                                                          (short)0, czero, false, false);
        sacc = __builtin_amdgcn_wmma_f32_16x16x32_f16(false, a3[1], false, bW3[1],
                                                      (short)0, sacc, false, false);

        // bias + lrelu, store s values (head columns 0..3) into softmax buffer
        if (nsel < 4) {
            const int mb = hi16 * 8;
            const float bias = b3[nsel];
#pragma unroll
            for (int rr = 0; rr < 8; ++rr)
                s_lds[(mt * 16 + rr + mb) * 4 + nsel] = lrelu(sacc[rr] + bias);
        }
    }
    __syncthreads();

    // --- Fused softmax over the 1024 s values of this group ---
    float v[4];
#pragma unroll
    for (int c = 0; c < 4; ++c) v[c] = s_lds[tid * 4 + c];
    float lmax = fmaxf(fmaxf(v[0], v[1]), fmaxf(v[2], v[3]));
    red_lds[tid] = lmax;
    __syncthreads();
    for (int s2 = 128; s2 > 0; s2 >>= 1) {
        if (tid < s2) red_lds[tid] = fmaxf(red_lds[tid], red_lds[tid + s2]);
        __syncthreads();
    }
    const float gmax = red_lds[0];
    __syncthreads();
    float lsum = 0.f;
#pragma unroll
    for (int c = 0; c < 4; ++c) { v[c] = __expf(v[c] - gmax); lsum += v[c]; }
    red_lds[tid] = lsum;
    __syncthreads();
    for (int s2 = 128; s2 > 0; s2 >>= 1) {
        if (tid < s2) red_lds[tid] += red_lds[tid + s2];
        __syncthreads();
    }
    const float inv = 1.0f / red_lds[0];
    float* op = out + (size_t)i * 4096 + (size_t)r * 1024 + (size_t)tid * 4;
#pragma unroll
    for (int c = 0; c < 4; ++c) op[c] = v[c] * inv;
}

extern "C" void kernel_launch(void* const* d_in, const int* in_sizes, int n_in,
                              void* d_out, int out_size, void* d_ws, size_t ws_size,
                              hipStream_t stream) {
    const float* x  = (const float*)d_in[0];
    const float* W1 = (const float*)d_in[1];
    const float* b1 = (const float*)d_in[2];
    const float* W2 = (const float*)d_in[3];
    const float* b2 = (const float*)d_in[4];
    const float* W3 = (const float*)d_in[5];
    const float* b3 = (const float*)d_in[6];
    float* out = (float*)d_out;

    float* hi  = (float*)d_ws;            // 1024*64 floats
    float* hjb = hi + NTOK * WIDTH;       // 1024*64 floats (b1 folded in)

    prep_kernel<<<(NTOK * WIDTH) / 256, 256, 0, stream>>>(x, W1, b1, hi, hjb);
    fused_attn_kernel<<<NTOK * 4, 256, 0, stream>>>(hi, hjb, W2, b2, W3, b3, out);
}
